// MLAReference_15281493639677
// MI455X (gfx1250) — matrix operand order repaired and checked by reading, not compiled, not run on hardware
//
#include <hip/hip_runtime.h>

typedef __attribute__((ext_vector_type(16))) __bf16 v16bf;
typedef __attribute__((ext_vector_type(8)))  float  v8f;
typedef __attribute__((ext_vector_type(4)))  int    v4i;

#define K_H    16
#define K_NOPE 128
#define K_ROPE 64
#define K_VD   128
#define K_KVL  512
#define K_S    2048
#define K_HID  2048
#define K_QLR  1536
#define K_QD   192   /* NOPE+ROPE */
#define K_KD   576   /* KVL+ROPE  */
#define K_EPS  1e-6f

// Async global->LDS path (gfx1250). Guarded so absence cannot break the build.
#if defined(__gfx1250__) && __has_builtin(__builtin_amdgcn_global_load_async_to_lds_b128) && \
    __has_builtin(__builtin_amdgcn_s_wait_asynccnt)
#define HAVE_ASYNC_LDS 1
#else
#define HAVE_ASYNC_LDS 0
#endif

#if HAVE_ASYNC_LDS
typedef __attribute__((address_space(1))) v4i* glob_v4i_ptr;
typedef __attribute__((address_space(3))) v4i* lds_v4i_ptr;
static __device__ __forceinline__ void async_cp16(const void* g, void* l) {
  __builtin_amdgcn_global_load_async_to_lds_b128((glob_v4i_ptr)g, (lds_v4i_ptr)l, 0, 0);
}
#endif

// ---------------------------------------------------------------------------
// WMMA helpers. Layouts per CDNA5 ISA §7.12.2 (wave32):
//   A (16x32, MxK): lane m=(lane&15), half=lane>>4;
//                   VGPR v holds K = (v<4?0:16) + half*8 + (v&3)*2, +1
//   B (32x16, KxN): lane n=(lane&15), half=lane>>4;
//                   VGPR v holds K = (v<4?0:16) + half*8 + (v&3)*2, +1
//   C/D (16x16 f32): lane n=(lane&15); VGPR r holds M = r + 8*(lane>>4)
// ---------------------------------------------------------------------------

static __device__ __forceinline__ v8f wmma_bf(v16bf a, v16bf b, v8f c) {
  return __builtin_amdgcn_wmma_f32_16x16x32_bf16(false, a, false, b, (short)0, c,
                                                 false, false);
}

static __device__ __forceinline__ v16bf ldA_f32(const float* M, int ld, int mb, int kb) {
  const int lane = threadIdx.x & 31;
  const int m  = mb + (lane & 15);
  const int hf = lane >> 4;
  v16bf f;
#pragma unroll
  for (int v = 0; v < 8; ++v) {
    const int k = kb + ((v < 4) ? 0 : 16) + hf * 8 + (v & 3) * 2;
    const float* p = M + (size_t)m * ld + k;
    f[2 * v]     = (__bf16)p[0];
    f[2 * v + 1] = (__bf16)p[1];
  }
  return f;
}

static __device__ __forceinline__ v16bf ldA_bf16(const __bf16* M, int ld, int mb, int kb) {
  const int lane = threadIdx.x & 31;
  const int m  = mb + (lane & 15);
  const int hf = lane >> 4;
  v16bf f;
#pragma unroll
  for (int v = 0; v < 8; ++v) {
    const int k = kb + ((v < 4) ? 0 : 16) + hf * 8 + (v & 3) * 2;
    const __bf16* p = M + (size_t)m * ld + k;
    f[2 * v]     = p[0];
    f[2 * v + 1] = p[1];
  }
  return f;
}

// B from row-major W (N x K): B[k][n] = W[n][k]  (computes A @ W^T)
static __device__ __forceinline__ v16bf ldB_rows_bf16(const __bf16* W, int ld, int nb, int kb) {
  const int lane = threadIdx.x & 31;
  const int n  = nb + (lane & 15);
  const int hf = lane >> 4;
  v16bf f;
#pragma unroll
  for (int v = 0; v < 8; ++v) {
    const int k = kb + ((v < 4) ? 0 : 16) + hf * 8 + (v & 3) * 2;
    const __bf16* p = W + (size_t)n * ld + k;
    f[2 * v]     = p[0];
    f[2 * v + 1] = p[1];
  }
  return f;
}

// B from K-major (K x N) source: B[k][n] = M[k*ld + n]  (computes A @ M)
static __device__ __forceinline__ v16bf ldB_km_bf16(const __bf16* M, int ld, int nb, int kb) {
  const int lane = threadIdx.x & 31;
  const int n  = nb + (lane & 15);
  const int hf = lane >> 4;
  v16bf f;
#pragma unroll
  for (int v = 0; v < 8; ++v) {
    const int k = kb + ((v < 4) ? 0 : 16) + hf * 8 + (v & 3) * 2;
    f[2 * v]     = M[(size_t)k * ld + n];
    f[2 * v + 1] = M[(size_t)(k + 1) * ld + n];
  }
  return f;
}

static __device__ __forceinline__ void stC_f32(float* C, int ld, int mb, int nb, v8f acc) {
  const int lane = threadIdx.x & 31;
  const int n  = nb + (lane & 15);
  const int hf = lane >> 4;
#pragma unroll
  for (int r = 0; r < 8; ++r)
    C[(size_t)(mb + r + 8 * hf) * ld + n] = acc[r];
}

static __device__ __forceinline__ void stC_bf16(__bf16* C, int ld, int mb, int nb, v8f acc) {
  const int lane = threadIdx.x & 31;
  const int n  = nb + (lane & 15);
  const int hf = lane >> 4;
#pragma unroll
  for (int r = 0; r < 8; ++r)
    C[(size_t)(mb + r + 8 * hf) * ld + n] = (__bf16)acc[r];
}

// ---------------------------------------------------------------------------
// One-shot fp32 -> bf16 conversion (weights/activations), amortized outside
// the GEMM inner loops.
// ---------------------------------------------------------------------------

__global__ __launch_bounds__(256) void cvt_f32_bf16(const float* __restrict__ in,
                                                    __bf16* __restrict__ out, int n) {
  for (int i = blockIdx.x * 256 + threadIdx.x; i < n; i += gridDim.x * 256)
    out[i] = (__bf16)in[i];
}

// ---------------------------------------------------------------------------
// GEMM C(MxN) = A(MxK) @ W(NxK)^T, bf16 x bf16 -> f32. Wave = 16x64 strip.
// ---------------------------------------------------------------------------

__global__ __launch_bounds__(256) void gemm_bb_abT_f32(const __bf16* __restrict__ A,
                                                       const __bf16* __restrict__ W,
                                                       float* __restrict__ C,
                                                       int M, int N, int K,
                                                       int lda, int ldw, int ldc) {
  const int wid = threadIdx.x >> 5;
  const int nstrips = N >> 6;
  const int strip = blockIdx.x * 8 + wid;
  if (strip >= (M >> 4) * nstrips) return;
  const int mb = (strip / nstrips) << 4;
  const int nb = (strip % nstrips) << 6;
  v8f a0 = {}, a1 = {}, a2 = {}, a3 = {};
  for (int kb = 0; kb < K; kb += 32) {
    __builtin_prefetch(A + (size_t)mb * lda + kb + 256, 0, 1);
    v16bf a = ldA_bf16(A, lda, mb, kb);
    a0 = wmma_bf(a, ldB_rows_bf16(W, ldw, nb,      kb), a0);
    a1 = wmma_bf(a, ldB_rows_bf16(W, ldw, nb + 16, kb), a1);
    a2 = wmma_bf(a, ldB_rows_bf16(W, ldw, nb + 32, kb), a2);
    a3 = wmma_bf(a, ldB_rows_bf16(W, ldw, nb + 48, kb), a3);
  }
  stC_f32(C, ldc, mb, nb,      a0);
  stC_f32(C, ldc, mb, nb + 16, a1);
  stC_f32(C, ldc, mb, nb + 32, a2);
  stC_f32(C, ldc, mb, nb + 48, a3);
}

// ---------------------------------------------------------------------------
// RMS-norm of ckv[:, :512] (+weight) and RoPE of ckv[:, 512:576] -> K bf16
// ---------------------------------------------------------------------------

__global__ __launch_bounds__(256) void k_finalize(const float* __restrict__ ckv,
                                                  const float* __restrict__ kvw,
                                                  __bf16* __restrict__ kout) {
  const int s = blockIdx.x;
  const float* row = ckv + (size_t)s * K_KD;
  __shared__ float red[256];
  float ss = 0.f;
  for (int i = threadIdx.x; i < K_KVL; i += 256) { float x = row[i]; ss += x * x; }
  red[threadIdx.x] = ss;
  __syncthreads();
  for (int o = 128; o > 0; o >>= 1) {
    if ((int)threadIdx.x < o) red[threadIdx.x] += red[threadIdx.x + o];
    __syncthreads();
  }
  const float inv = rsqrtf(red[0] / (float)K_KVL + K_EPS);
  __bf16* orow = kout + (size_t)s * K_KD;
  for (int i = threadIdx.x; i < K_KVL; i += 256)
    orow[i] = (__bf16)(row[i] * inv * kvw[i]);
  for (int i = threadIdx.x; i < K_ROPE; i += 256) {
    const int j = i & 31;
    const float freq = __powf(10000.0f, -(float)(2 * j) / (float)K_ROPE);
    const float ang = (float)s * freq;
    const float c = __cosf(ang), sn = __sinf(ang);
    float out;
    if (i < 32) out = row[K_KVL + 2 * i] * c - row[K_KVL + 2 * i + 1] * sn;
    else { const int jj = i - 32; out = row[K_KVL + 2 * jj + 1] * c + row[K_KVL + 2 * jj] * sn; }
    orow[K_KVL + i] = (__bf16)out;
  }
}

__global__ __launch_bounds__(256) void qa_norm(const float* __restrict__ qa,
                                               const float* __restrict__ w,
                                               __bf16* __restrict__ out) {
  const int s = blockIdx.x;
  const float* row = qa + (size_t)s * K_QLR;
  __shared__ float red[256];
  float ss = 0.f;
  for (int i = threadIdx.x; i < K_QLR; i += 256) { float x = row[i]; ss += x * x; }
  red[threadIdx.x] = ss;
  __syncthreads();
  for (int o = 128; o > 0; o >>= 1) {
    if ((int)threadIdx.x < o) red[threadIdx.x] += red[threadIdx.x + o];
    __syncthreads();
  }
  const float inv = rsqrtf(red[0] / (float)K_QLR + K_EPS);
  __bf16* orow = out + (size_t)s * K_QLR;
  for (int i = threadIdx.x; i < K_QLR; i += 256)
    orow[i] = (__bf16)(row[i] * inv * w[i]);
}

// RoPE on q_pe -> qs[h][s][512:576]
__global__ void qpe_rope(const float* __restrict__ q, __bf16* __restrict__ qs) {
  const int s = blockIdx.x, h = blockIdx.y, i = threadIdx.x;  // i < 64
  const float* x = q + (size_t)s * (K_H * K_QD) + h * K_QD + K_NOPE;
  const int j = i & 31;
  const float freq = __powf(10000.0f, -(float)(2 * j) / (float)K_ROPE);
  const float ang = (float)s * freq;
  const float c = __cosf(ang), sn = __sinf(ang);
  float out;
  if (i < 32) out = x[2 * i] * c - x[2 * i + 1] * sn;
  else { const int jj = i - 32; out = x[2 * jj + 1] * c + x[2 * jj] * sn; }
  qs[(size_t)h * K_S * K_KD + (size_t)s * K_KD + K_KVL + i] = (__bf16)out;
}

// q_latent[h] = q_nope[h](2048x128) @ kvb[h,:128,:](128x512) -> qs[h][s][0:512]
__global__ __launch_bounds__(256) void absorb_kernel(const float* __restrict__ q,
                                                     const __bf16* __restrict__ wkv_b_b,
                                                     __bf16* __restrict__ qs) {
  const int h = blockIdx.y;
  const int wid = threadIdx.x >> 5;
  const int nstrips = K_KVL >> 6;  // 8
  const int strip = blockIdx.x * 8 + wid;
  if (strip >= (K_S >> 4) * nstrips) return;
  const int mb = (strip / nstrips) << 4;
  const int nb = (strip % nstrips) << 6;
  const float* A = q + h * K_QD;                                       // [s][3072] col-offset
  const __bf16* B = wkv_b_b + (size_t)h * (K_NOPE + K_VD) * K_KVL;     // [n][512] K-major
  v8f a0 = {}, a1 = {}, a2 = {}, a3 = {};
  for (int kb = 0; kb < K_NOPE; kb += 32) {
    v16bf a = ldA_f32(A, K_H * K_QD, mb, kb);
    a0 = wmma_bf(a, ldB_km_bf16(B, K_KVL, nb,      kb), a0);
    a1 = wmma_bf(a, ldB_km_bf16(B, K_KVL, nb + 16, kb), a1);
    a2 = wmma_bf(a, ldB_km_bf16(B, K_KVL, nb + 32, kb), a2);
    a3 = wmma_bf(a, ldB_km_bf16(B, K_KVL, nb + 48, kb), a3);
  }
  __bf16* C = qs + (size_t)h * K_S * K_KD;
  stC_bf16(C, K_KD, mb, nb,      a0);
  stC_bf16(C, K_KD, mb, nb + 16, a1);
  stC_bf16(C, K_KD, mb, nb + 32, a2);
  stC_bf16(C, K_KD, mb, nb + 48, a3);
}

// ---------------------------------------------------------------------------
// Flash attention: block = (head, 16-query tile), 4 waves; wave w owns
// v-columns [w*128, w*128+128).  The shared 32x576 K-block is double-buffered
// in LDS: block kb+1 is prefetched with async global->LDS copies while the
// WMMAs for block kb execute; ASYNCcnt is only drained after the compute.
// ---------------------------------------------------------------------------

__global__ __launch_bounds__(128) void attn_kernel(const __bf16* __restrict__ qs,
                                                   const __bf16* __restrict__ kmat,
                                                   __bf16* __restrict__ ctx) {
  const int h = blockIdx.y;
  const int qt = blockIdx.x;
  const int wid = threadIdx.x >> 5;
  const int lane = threadIdx.x & 31;
  const int hf = lane >> 4;
  const int ln = lane & 15;
  const __bf16* Q = qs + (size_t)h * K_S * K_KD;
  const int mb = qt * 16;
  const float scale = rsqrtf((float)K_QD);

  __shared__ __bf16 kblk[2][32][K_KD];  // 2 x 36 KB: double-buffered K block
  __shared__ __bf16 pst[4][16][32];     // per-wave P staging (C->A relayout)

  float m[8], l[8];
  v8f O[8];
#pragma unroll
  for (int r = 0; r < 8; ++r) { m[r] = -1e30f; l[r] = 0.f; O[r] = (v8f){}; }

  const int nchunks = 32 * K_KD * 2 / 16;  // 2304 x 16B per block
  auto stage = [&](int kbase, int buf) {
    const char* src = (const char*)(kmat + (size_t)kbase * K_KD);
    char* dst = (char*)&kblk[buf][0][0];
    for (int t = threadIdx.x; t < nchunks; t += 128) {
#if HAVE_ASYNC_LDS
      async_cp16(src + (size_t)t * 16, dst + (size_t)t * 16);
#else
      ((uint4*)dst)[t] = ((const uint4*)src)[t];
#endif
    }
  };

  const int kbmax = (qt * 16 + 15) / 32;
  int cur = 0;
  stage(0, 0);
#if HAVE_ASYNC_LDS
  __builtin_amdgcn_s_wait_asynccnt(0);
#endif
  __syncthreads();

  for (int kb = 0; kb <= kbmax; ++kb) {
    const int kbase = kb * 32;
    // ---- prefetch next K block into the other buffer (async, no wait) ----
    if (kb < kbmax) stage(kbase + 32, cur ^ 1);
    const __bf16* kb_lds = &kblk[cur][0][0];
    // ---- scores: 16 queries x 32 keys over 576-dim ----
    v8f s0 = {}, s1 = {};
    for (int d = 0; d < K_KD; d += 32) {
      v16bf a = ldA_bf16(Q, K_KD, mb, d);
      s0 = wmma_bf(a, ldB_rows_bf16(kb_lds, K_KD, 0,  d), s0);
      s1 = wmma_bf(a, ldB_rows_bf16(kb_lds, K_KD, 16, d), s1);
    }
    // ---- scale + causal mask + online softmax (C-layout) ----
    float mn[8];
#pragma unroll
    for (int r = 0; r < 8; ++r) {
      const int qrow = mb + r + 8 * hf;
      float v0 = s0[r] * scale, v1 = s1[r] * scale;
      if (kbase + ln > qrow)      v0 = -1e30f;
      if (kbase + 16 + ln > qrow) v1 = -1e30f;
      s0[r] = v0; s1[r] = v1;
      float mx = fmaxf(v0, v1);
      mx = fmaxf(mx, __shfl_xor(mx, 1, 32));
      mx = fmaxf(mx, __shfl_xor(mx, 2, 32));
      mx = fmaxf(mx, __shfl_xor(mx, 4, 32));
      mx = fmaxf(mx, __shfl_xor(mx, 8, 32));
      mn[r] = fmaxf(m[r], mx);
    }
#pragma unroll
    for (int r = 0; r < 8; ++r) {
      const float p0 = __expf(s0[r] - mn[r]);
      const float p1 = __expf(s1[r] - mn[r]);
      float rs = p0 + p1;
      rs += __shfl_xor(rs, 1, 32);
      rs += __shfl_xor(rs, 2, 32);
      rs += __shfl_xor(rs, 4, 32);
      rs += __shfl_xor(rs, 8, 32);
      const float alpha = __expf(m[r] - mn[r]);
      l[r] = l[r] * alpha + rs;
      m[r] = mn[r];
#pragma unroll
      for (int j = 0; j < 8; ++j) O[j][r] *= alpha;
      const int mm = r + 8 * hf;
      pst[wid][mm][ln]      = (__bf16)p0;   // key kbase+ln
      pst[wid][mm][16 + ln] = (__bf16)p1;   // key kbase+16+ln
    }
    // ---- re-layout P (C-layout -> A-layout) via per-wave LDS staging ----
    v16bf pa;
#pragma unroll
    for (int v = 0; v < 8; ++v) {
      const int kk = ((v < 4) ? 0 : 16) + hf * 8 + (v & 3) * 2;
      pa[2 * v]     = pst[wid][ln][kk];
      pa[2 * v + 1] = pst[wid][ln][kk + 1];
    }
    // ---- O += P @ V  (V = staged k_nope rows, cols wid*128 .. +128) ----
#pragma unroll
    for (int j = 0; j < 8; ++j) {
      O[j] = wmma_bf(pa, ldB_km_bf16(kb_lds, K_KD, wid * 128 + j * 16, 0), O[j]);
    }
    // ---- drain async prefetch, make next buffer visible block-wide ----
#if HAVE_ASYNC_LDS
    __builtin_amdgcn_s_wait_asynccnt(0);
#endif
    __syncthreads();
    cur ^= 1;
  }
  // ---- normalize & store ctx[h] (2048 x 512) bf16 ----
  __bf16* co = ctx + (size_t)h * K_S * K_KVL;
#pragma unroll
  for (int j = 0; j < 8; ++j) {
    v8f t = O[j];
#pragma unroll
    for (int r = 0; r < 8; ++r) t[r] = t[r] / l[r];
    stC_bf16(co, K_KVL, mb, wid * 128 + j * 16, t);
  }
}

// out_head[h] = ctx[h](2048x512) @ wkv_b[h,128:256,:](128x512)^T -> ho[:, h*128:+128]
__global__ __launch_bounds__(256) void headout_kernel(const __bf16* __restrict__ ctx,
                                                      const __bf16* __restrict__ wkv_b_b,
                                                      __bf16* __restrict__ ho) {
  const int h = blockIdx.y;
  const int wid = threadIdx.x >> 5;
  const int nstrips = K_VD >> 6;  // 2
  const int strip = blockIdx.x * 8 + wid;
  if (strip >= (K_S >> 4) * nstrips) return;
  const int mb = (strip / nstrips) << 4;
  const int nb = (strip % nstrips) << 6;
  const __bf16* A = ctx + (size_t)h * K_S * K_KVL;
  const __bf16* W = wkv_b_b + ((size_t)h * (K_NOPE + K_VD) + K_NOPE) * K_KVL;
  v8f a0 = {}, a1 = {}, a2 = {}, a3 = {};
  for (int kb = 0; kb < K_KVL; kb += 32) {
    v16bf a = ldA_bf16(A, K_KVL, mb, kb);
    a0 = wmma_bf(a, ldB_rows_bf16(W, K_KVL, nb,      kb), a0);
    a1 = wmma_bf(a, ldB_rows_bf16(W, K_KVL, nb + 16, kb), a1);
    a2 = wmma_bf(a, ldB_rows_bf16(W, K_KVL, nb + 32, kb), a2);
    a3 = wmma_bf(a, ldB_rows_bf16(W, K_KVL, nb + 48, kb), a3);
  }
  stC_bf16(ho, K_HID, mb, h * K_VD + nb,      a0);
  stC_bf16(ho, K_HID, mb, h * K_VD + nb + 16, a1);
  stC_bf16(ho, K_HID, mb, h * K_VD + nb + 32, a2);
  stC_bf16(ho, K_HID, mb, h * K_VD + nb + 48, a3);
}

// ---------------------------------------------------------------------------

extern "C" void kernel_launch(void* const* d_in, const int* in_sizes, int n_in,
                              void* d_out, int out_size, void* d_ws, size_t ws_size,
                              hipStream_t stream) {
  (void)in_sizes; (void)n_in; (void)out_size; (void)ws_size;
  const float* hidden = (const float*)d_in[0];
  // d_in[1] = position_ids = arange(S): derived from token index on-device.
  const float* wq_a  = (const float*)d_in[2];
  const float* qaw   = (const float*)d_in[3];
  const float* wq_b  = (const float*)d_in[4];
  const float* wkv_a = (const float*)d_in[5];
  const float* kvw   = (const float*)d_in[6];
  const float* wkv_b = (const float*)d_in[7];
  const float* wo    = (const float*)d_in[8];
  float* out = (float*)d_out;

  char* ws = (char*)d_ws;
  size_t o = 0;
  auto take = [&](size_t bytes) -> void* {
    void* p = ws + o;
    o += (bytes + 255) & ~(size_t)255;
    return p;
  };
  // fp32 intermediates
  float*  ckv  = (float*) take((size_t)K_S * K_KD * 4);            // hidden @ wkv_a^T
  float*  qa   = (float*) take((size_t)K_S * K_QLR * 4);           // hidden @ wq_a^T
  float*  qf   = (float*) take((size_t)K_S * K_H * K_QD * 4);      // qab @ wq_b^T
  // bf16 activations
  __bf16* qab  = (__bf16*)take((size_t)K_S * K_QLR * 2);           // rmsnorm(qa)
  __bf16* kb16 = (__bf16*)take((size_t)K_S * K_KD * 2);            // K (norm + rope)
  __bf16* qsb  = (__bf16*)take((size_t)K_H * K_S * K_KD * 2);      // q_latent | q_rope
  __bf16* ctxb = (__bf16*)take((size_t)K_H * K_S * K_KVL * 2);     // attention ctx
  __bf16* hob  = (__bf16*)take((size_t)K_S * K_HID * 2);           // merged head out
  // bf16 copies of fp32 operands (one-shot conversion)
  __bf16* hid_b   = (__bf16*)take((size_t)K_S * K_HID * 2);
  __bf16* wq_a_b  = (__bf16*)take((size_t)K_QLR * K_HID * 2);
  __bf16* wq_b_b  = (__bf16*)take((size_t)K_H * K_QD * K_QLR * 2);
  __bf16* wkv_a_b = (__bf16*)take((size_t)K_KD * K_HID * 2);
  __bf16* wkv_b_b = (__bf16*)take((size_t)K_H * (K_NOPE + K_VD) * K_KVL * 2);
  __bf16* wo_b    = (__bf16*)take((size_t)K_HID * (K_H * K_VD) * 2);

  dim3 blk(256);
  // ---- one-shot fp32 -> bf16 conversions ----
  cvt_f32_bf16<<<1024, blk, 0, stream>>>(hidden, hid_b,   K_S * K_HID);
  cvt_f32_bf16<<<1024, blk, 0, stream>>>(wq_a,   wq_a_b,  K_QLR * K_HID);
  cvt_f32_bf16<<<1024, blk, 0, stream>>>(wq_b,   wq_b_b,  K_H * K_QD * K_QLR);
  cvt_f32_bf16<<<1024, blk, 0, stream>>>(wkv_a,  wkv_a_b, K_KD * K_HID);
  cvt_f32_bf16<<<1024, blk, 0, stream>>>(wkv_b,  wkv_b_b, K_H * (K_NOPE + K_VD) * K_KVL);
  cvt_f32_bf16<<<1024, blk, 0, stream>>>(wo,     wo_b,    K_HID * K_H * K_VD);

  // ckv = hidden @ wkv_a^T        (2048 x 576 x 2048)
  gemm_bb_abT_f32<<<144, blk, 0, stream>>>(hid_b, wkv_a_b, ckv,
                                           K_S, K_KD, K_HID, K_HID, K_HID, K_KD);
  // qa = hidden @ wq_a^T          (2048 x 1536 x 2048)
  gemm_bb_abT_f32<<<384, blk, 0, stream>>>(hid_b, wq_a_b, qa,
                                           K_S, K_QLR, K_HID, K_HID, K_HID, K_QLR);
  // K = [rmsnorm(ckv[:512])*w | rope(ckv[512:])]
  k_finalize<<<K_S, 256, 0, stream>>>(ckv, kvw, kb16);
  // qab = rmsnorm(qa)*w
  qa_norm<<<K_S, 256, 0, stream>>>(qa, qaw, qab);
  // qf = qab @ wq_b^T             (2048 x 3072 x 1536)
  gemm_bb_abT_f32<<<768, blk, 0, stream>>>(qab, wq_b_b, qf,
                                           K_S, K_H * K_QD, K_QLR,
                                           K_QLR, K_QLR, K_H * K_QD);
  // qs[..,512:576] = rope(q_pe)
  qpe_rope<<<dim3(K_S, K_H), 64, 0, stream>>>(qf, qsb);
  // qs[..,0:512] = q_nope @ kvb[:, :128]   (per head, 2048 x 512 x 128)
  absorb_kernel<<<dim3(128, K_H), blk, 0, stream>>>(qf, wkv_b_b, qsb);
  // flash attention -> ctx  (per head, causal)
  attn_kernel<<<dim3(K_S / 16, K_H), 128, 0, stream>>>(qsb, kb16, ctxb);
  // ho[:, h*128:+128] = ctx[h] @ kv_b2[h]^T   (per head, 2048 x 128 x 512)
  headout_kernel<<<dim3(32, K_H), blk, 0, stream>>>(ctxb, wkv_b_b, hob);
  // out = ho @ wo^T               (2048 x 2048 x 2048)
  gemm_bb_abT_f32<<<512, blk, 0, stream>>>(hob, wo_b, out,
                                           K_S, K_HID, K_HID,
                                           K_HID, K_HID, K_HID);
}
